// MoELayer_55387898249456
// MI455X (gfx1250) — matrix operand order repaired and checked by reading, not compiled, not run on hardware
//
#include <hip/hip_runtime.h>
#include <hip/hip_bf16.h>
#include <math.h>

// ---------------- problem dims ----------------
constexpr int NB = 4, NT = 2048, ND = 1024, NF = 4096, NE = 8, NK = 2;
constexpr int NTOK  = NB * NT;            // 8192 tokens
constexpr int NPAIR = NTOK * NK;          // 16384 (token, expert) pairs
constexpr int MAXT  = NPAIR / 128 + NE;   // 136 worst-case M tiles across experts

// ---------------- WMMA types ----------------
typedef __attribute__((ext_vector_type(16))) __bf16 bf16x16;
typedef __attribute__((ext_vector_type(8)))  float  f32x8;

union FragAB { bf16x16 v; uint4 q[2]; };

// ---------------- async global->LDS copy (CDNA5), with sync fallback ----------
#if defined(__has_builtin)
#if __has_builtin(__builtin_amdgcn_global_load_async_to_lds_b128) && \
    __has_builtin(__builtin_amdgcn_s_wait_asynccnt)
#define USE_ASYNC_LDS 1
#else
#define USE_ASYNC_LDS 0
#endif
#else
#define USE_ASYNC_LDS 0
#endif

typedef int v4i_a __attribute__((vector_size(16)));

__device__ inline void cp_async16(void* lds, const void* g) {
#if USE_ASYNC_LDS
    __builtin_amdgcn_global_load_async_to_lds_b128(
        (__attribute__((address_space(1))) v4i_a*)(g),
        (__attribute__((address_space(3))) v4i_a*)(lds), 0, 0);
#else
    *(uint4*)lds = *(const uint4*)g;
#endif
}
__device__ inline void cp_wait_all() {
#if USE_ASYNC_LDS
    __builtin_amdgcn_s_wait_asynccnt(0);
#endif
}

// ---------------- workspace layout (bytes), total ~286 MB ----------------
constexpr size_t WS_COUNTS = 0;                                  // 8  int
constexpr size_t WS_OFFS   = 64;                                 // 9  int
constexpr size_t WS_CURS   = 128;                                // 8  int
constexpr size_t WS_ESUM   = 192;                                // 8  float
constexpr size_t WS_TOKE   = 256;                                // NPAIR int
constexpr size_t WS_TOKW   = WS_TOKE + (size_t)NPAIR * 4;        // NPAIR float
constexpr size_t WS_BTOK   = WS_TOKW + (size_t)NPAIR * 4;        // NPAIR int
constexpr size_t WS_BW     = WS_BTOK + (size_t)NPAIR * 4;        // NPAIR float
constexpr size_t WS_XB     = (WS_BW + (size_t)NPAIR * 4 + 255) & ~(size_t)255; // N*D bf16
constexpr size_t WS_MID    = WS_XB  + (size_t)NTOK * ND * 2;     // NPAIR*F bf16
constexpr size_t WS_W1T    = WS_MID + (size_t)NPAIR * NF * 2;    // E*F*D bf16 (W1 transposed)
constexpr size_t WS_W2T    = WS_W1T + (size_t)NE * ND * NF * 2;  // E*D*F bf16 (W2 transposed)

// ---------------- gate: logits, top-2, softmax ----------------
__global__ __launch_bounds__(256)
void gate_kernel(const float* __restrict__ x, const float* __restrict__ Wg,
                 int* __restrict__ toke, float* __restrict__ tokw,
                 int* __restrict__ counts) {
    const int lane = threadIdx.x & 31;
    const int n = (int)((blockIdx.x * blockDim.x + threadIdx.x) >> 5);
    if (n >= NTOK) return;

    float acc[NE] = {};
    for (int d = lane; d < ND; d += 32) {
        const float xv = x[(size_t)n * ND + d];
        const float4* w4 = (const float4*)(Wg + (size_t)d * NE);
        const float4 a = w4[0], b = w4[1];
        acc[0] += xv * a.x; acc[1] += xv * a.y; acc[2] += xv * a.z; acc[3] += xv * a.w;
        acc[4] += xv * b.x; acc[5] += xv * b.y; acc[6] += xv * b.z; acc[7] += xv * b.w;
    }
    #pragma unroll
    for (int e = 0; e < NE; ++e)
        #pragma unroll
        for (int off = 16; off > 0; off >>= 1)
            acc[e] += __shfl_xor(acc[e], off, 32);

    if (lane == 0) {
        int i0 = 0; float s0 = acc[0];
        for (int e = 1; e < NE; ++e) if (acc[e] > s0) { s0 = acc[e]; i0 = e; }
        int i1 = -1; float s1 = -3.4e38f;
        for (int e = 0; e < NE; ++e) if (e != i0 && acc[e] > s1) { s1 = acc[e]; i1 = e; }
        const float ex = expf(s1 - s0);                // s0 >= s1
        const float w0 = 1.0f / (1.0f + ex);
        const float w1 = ex / (1.0f + ex);
        toke[n * 2]     = i0;  toke[n * 2 + 1] = i1;
        tokw[n * 2]     = w0;  tokw[n * 2 + 1] = w1;
        atomicAdd(&counts[i0], 1);
        atomicAdd(&counts[i1], 1);
    }
}

// ---------------- tiny prefix sum over 8 experts ----------------
__global__ void prefix_kernel(const int* __restrict__ counts,
                              int* __restrict__ offs, int* __restrict__ curs) {
    int a = 0;
    for (int e = 0; e < NE; ++e) { offs[e] = a; curs[e] = a; a += counts[e]; }
    offs[NE] = a;
}

// ---------------- scatter tokens into expert buckets ----------------
__global__ __launch_bounds__(256)
void scatter_kernel(const int* __restrict__ toke, const float* __restrict__ tokw,
                    int* __restrict__ curs, int* __restrict__ btok,
                    float* __restrict__ bw) {
    const int n = (int)(blockIdx.x * blockDim.x + threadIdx.x);
    if (n >= NTOK) return;
    #pragma unroll
    for (int k = 0; k < NK; ++k) {
        const int e = toke[n * 2 + k];
        const int pos = atomicAdd(&curs[e], 1);
        btok[pos] = n;
        bw[pos] = tokw[n * 2 + k];
    }
}

// ---------------- cast x (fp32) -> bf16 once ----------------
__global__ __launch_bounds__(256)
void castx_kernel(const float* __restrict__ x, __bf16* __restrict__ xb) {
    const size_t i = (size_t)blockIdx.x * blockDim.x + threadIdx.x; // N*D/4 threads
    const float4 v = ((const float4*)x)[i];
    union { __bf16 h[4]; uint2 u; } o;
    o.h[0] = (__bf16)v.x; o.h[1] = (__bf16)v.y;
    o.h[2] = (__bf16)v.z; o.h[3] = (__bf16)v.w;
    ((uint2*)xb)[i] = o.u;
}

// ---------------- weight transpose+cast: [E][R][C] fp32 -> [E][C][R] bf16 -----
__global__ __launch_bounds__(256)
void wtrans_kernel(const float* __restrict__ W, __bf16* __restrict__ Wt,
                   int R, int C) {
    __shared__ float tile[32][33];
    const int e  = (int)blockIdx.z;
    const int c0 = (int)blockIdx.x * 32, r0 = (int)blockIdx.y * 32;
    const int tx = threadIdx.x & 31, ty = threadIdx.x >> 5;   // 32 x 8
    const float* src = W  + (size_t)e * R * C;
    __bf16*      dst = Wt + (size_t)e * R * C;
    #pragma unroll
    for (int j = 0; j < 4; ++j)
        tile[ty + 8 * j][tx] = src[(size_t)(r0 + ty + 8 * j) * C + c0 + tx];
    __syncthreads();
    #pragma unroll
    for (int j = 0; j < 4; ++j)
        dst[(size_t)(c0 + ty + 8 * j) * R + r0 + tx] = (__bf16)tile[tx][ty + 8 * j];
}

// map blockIdx.y -> (expert, bucket row range); uniform across block
__device__ inline bool find_tile(const int* counts, const int* offs, int mt,
                                 int& e, int& row0, int& rows) {
    int acc = 0;
    for (int i = 0; i < NE; ++i) {
        const int c = counts[i];
        const int t = (c + 127) >> 7;
        if (mt < acc + t) {
            const int lm = mt - acc;
            e = i;
            row0 = offs[i] + lm * 128;
            rows = c - lm * 128; if (rows > 128) rows = 128;
            return true;
        }
        acc += t;
    }
    return false;
}

// ---------------- up-proj: mid = silu(X_bucket @ W1[e]) ----------------
// block tile 128(M) x 128(N=F), K-step 32, double-buffered async LDS staging;
// 8 waves = 2x4, wave tile 64x32 = 4x2 wmma 16x16 tiles
__global__ __launch_bounds__(256)
void up_kernel(const __bf16* __restrict__ xb, const __bf16* __restrict__ W1t,
               const int* __restrict__ counts, const int* __restrict__ offs,
               const int* __restrict__ btok, __bf16* __restrict__ mid) {
    int e, row0, rows;
    if (!find_tile(counts, offs, (int)blockIdx.y, e, row0, rows)) return;
    const int f0 = (int)blockIdx.x * 128;
    const int tid = threadIdx.x, lane = tid & 31, wid = tid >> 5;
    const int wm = wid >> 2, wn = wid & 3;

    __shared__ __align__(16) unsigned char sA[2][128 * 80]; // [row][k] pitch 80B
    __shared__ __align__(16) unsigned char sB[2][128 * 80]; // [col][k] pitch 80B
    __shared__ int s_tok[128];

    // clamp OOB bucket rows to a valid row: their results are never stored
    if (tid < 128) s_tok[tid] = btok[row0 + ((tid < rows) ? tid : 0)];
    __syncthreads();

    const __bf16* Bsrc = W1t + (size_t)e * ND * NF;   // [F][D] bf16
    const int mrow = lane & 15, khalf = lane >> 4;
    f32x8 acc[4][2] = {};

    auto stage = [&](int p, int k0) {
        #pragma unroll
        for (int i = 0; i < 2; ++i) {
            const int c = tid * 2 + i;                // 0..511 16B chunks
            const int row = c >> 2, ko = (c & 3) * 8;
            cp_async16(sA[p] + row * 80 + ko * 2,
                       xb + (size_t)s_tok[row] * ND + k0 + ko);
        }
        #pragma unroll
        for (int i = 0; i < 2; ++i) {
            const int c = tid * 2 + i;
            const int row = c >> 2, ko = (c & 3) * 8;
            cp_async16(sB[p] + row * 80 + ko * 2,
                       Bsrc + (size_t)(f0 + row) * ND + k0 + ko);
        }
    };

    constexpr int KT = ND / 32;
    stage(0, 0);
    for (int kt = 0; kt < KT; ++kt) {
        const int p = kt & 1;
        cp_wait_all();
        __syncthreads();
        if (kt + 1 < KT) stage(p ^ 1, (kt + 1) * 32);

        FragAB a[4], b[2];
        #pragma unroll
        for (int mi = 0; mi < 4; ++mi) {
            const unsigned char* q = sA[p] + (wm * 64 + mi * 16 + mrow) * 80 + khalf * 16;
            a[mi].q[0] = *(const uint4*)q;
            a[mi].q[1] = *(const uint4*)(q + 32);
        }
        #pragma unroll
        for (int ni = 0; ni < 2; ++ni) {
            const unsigned char* q = sB[p] + (wn * 32 + ni * 16 + mrow) * 80 + khalf * 16;
            b[ni].q[0] = *(const uint4*)q;
            b[ni].q[1] = *(const uint4*)(q + 32);
        }
        #pragma unroll
        for (int mi = 0; mi < 4; ++mi)
            #pragma unroll
            for (int ni = 0; ni < 2; ++ni)
                acc[mi][ni] = __builtin_amdgcn_wmma_f32_16x16x32_bf16(
                    false, a[mi].v, false, b[ni].v, (short)0, acc[mi][ni], false, false);
    }

    // epilogue: silu -> mid (bf16)
    const int col = lane & 15, rh = lane >> 4;
    #pragma unroll
    for (int mi = 0; mi < 4; ++mi) {
        const int mBase = wm * 64 + mi * 16 + rh * 8;
        #pragma unroll
        for (int ni = 0; ni < 2; ++ni) {
            const int nCol = f0 + wn * 32 + ni * 16 + col;
            #pragma unroll
            for (int r = 0; r < 8; ++r) {
                const int m = mBase + r;
                if (m < rows) {
                    const float v = acc[mi][ni][r];
                    mid[(size_t)(row0 + m) * NF + nCol] = (__bf16)(v / (1.0f + expf(-v)));
                }
            }
        }
    }
}

// ---------------- down-proj: y += w * (mid @ W2[e]) ----------------
__global__ __launch_bounds__(256)
void down_kernel(const __bf16* __restrict__ mid, const __bf16* __restrict__ W2t,
                 const int* __restrict__ counts, const int* __restrict__ offs,
                 const int* __restrict__ btok, const float* __restrict__ bw,
                 float* __restrict__ y) {
    int e, row0, rows;
    if (!find_tile(counts, offs, (int)blockIdx.y, e, row0, rows)) return;
    const int d0 = (int)blockIdx.x * 128;
    const int tid = threadIdx.x, lane = tid & 31, wid = tid >> 5;
    const int wm = wid >> 2, wn = wid & 3;

    __shared__ __align__(16) unsigned char sA[2][128 * 80];
    __shared__ __align__(16) unsigned char sB[2][128 * 80];
    __shared__ int   s_tok[128];
    __shared__ float s_w[128];

    if (tid < 128) {
        const bool vld = tid < rows;
        s_tok[tid] = vld ? btok[row0 + tid] : 0;
        s_w[tid]   = vld ? bw[row0 + tid]   : 0.0f;
    }
    __syncthreads();

    const __bf16* Bsrc = W2t + (size_t)e * NF * ND;   // [D][F] bf16
    const int mrow = lane & 15, khalf = lane >> 4;
    f32x8 acc[4][2] = {};

    auto stage = [&](int p, int k0) {
        #pragma unroll
        for (int i = 0; i < 2; ++i) {
            const int c = tid * 2 + i;
            const int row = c >> 2, ko = (c & 3) * 8;
            const int rr = (row < rows) ? row : 0;    // clamp OOB rows
            cp_async16(sA[p] + row * 80 + ko * 2,
                       mid + (size_t)(row0 + rr) * NF + k0 + ko);
        }
        #pragma unroll
        for (int i = 0; i < 2; ++i) {
            const int c = tid * 2 + i;
            const int row = c >> 2, ko = (c & 3) * 8;
            cp_async16(sB[p] + row * 80 + ko * 2,
                       Bsrc + (size_t)(d0 + row) * NF + k0 + ko);
        }
    };

    constexpr int KT = NF / 32;
    stage(0, 0);
    for (int kt = 0; kt < KT; ++kt) {
        const int p = kt & 1;
        cp_wait_all();
        __syncthreads();
        if (kt + 1 < KT) stage(p ^ 1, (kt + 1) * 32);

        FragAB a[4], b[2];
        #pragma unroll
        for (int mi = 0; mi < 4; ++mi) {
            const unsigned char* q = sA[p] + (wm * 64 + mi * 16 + mrow) * 80 + khalf * 16;
            a[mi].q[0] = *(const uint4*)q;
            a[mi].q[1] = *(const uint4*)(q + 32);
        }
        #pragma unroll
        for (int ni = 0; ni < 2; ++ni) {
            const unsigned char* q = sB[p] + (wn * 32 + ni * 16 + mrow) * 80 + khalf * 16;
            b[ni].q[0] = *(const uint4*)q;
            b[ni].q[1] = *(const uint4*)(q + 32);
        }
        #pragma unroll
        for (int mi = 0; mi < 4; ++mi)
            #pragma unroll
            for (int ni = 0; ni < 2; ++ni)
                acc[mi][ni] = __builtin_amdgcn_wmma_f32_16x16x32_bf16(
                    false, a[mi].v, false, b[ni].v, (short)0, acc[mi][ni], false, false);
    }

    // epilogue: scale by gate weight, accumulate into y (2 commutative adds/elem)
    const int col = lane & 15, rh = lane >> 4;
    #pragma unroll
    for (int mi = 0; mi < 4; ++mi) {
        const int mBase = wm * 64 + mi * 16 + rh * 8;
        #pragma unroll
        for (int ni = 0; ni < 2; ++ni) {
            const int nCol = d0 + wn * 32 + ni * 16 + col;
            #pragma unroll
            for (int r = 0; r < 8; ++r) {
                const int m = mBase + r;
                if (m < rows)
                    atomicAdd(y + (size_t)s_tok[m] * ND + nCol, s_w[m] * acc[mi][ni][r]);
            }
        }
    }
}

// ---------------- deterministic per-expert weight sums ----------------
__global__ __launch_bounds__(256)
void esum_kernel(const int* __restrict__ toke, const float* __restrict__ tokw,
                 float* __restrict__ esum) {
    const int e = (int)blockIdx.x, t = threadIdx.x;
    __shared__ float red[256];
    float s = 0.0f;
    for (int i = t; i < NPAIR; i += 256)
        if (toke[i] == e) s += tokw[i];
    red[t] = s;
    __syncthreads();
    for (int w = 128; w > 0; w >>= 1) {
        if (t < w) red[t] += red[t + w];
        __syncthreads();
    }
    if (t == 0) esum[e] = red[0];
}

__global__ void aux_kernel(const float* __restrict__ esum, float* __restrict__ out) {
    float tot = 0.0f;
    for (int e = 0; e < NE; ++e) tot += esum[e];
    float a = 0.0f;
    for (int e = 0; e < NE; ++e) {
        const float l = esum[e] / tot;
        a += l * logf(l);
    }
    *out = a;
}

// ---------------- host launcher ----------------
extern "C" void kernel_launch(void* const* d_in, const int* in_sizes, int n_in,
                              void* d_out, int out_size, void* d_ws, size_t ws_size,
                              hipStream_t stream) {
    (void)in_sizes; (void)n_in; (void)out_size; (void)ws_size;
    const float* x  = (const float*)d_in[0];
    const float* Wg = (const float*)d_in[1];
    const float* W1 = (const float*)d_in[2];
    const float* W2 = (const float*)d_in[3];
    float* y = (float*)d_out;

    char* ws = (char*)d_ws;
    int*    counts = (int*)   (ws + WS_COUNTS);
    int*    offs   = (int*)   (ws + WS_OFFS);
    int*    curs   = (int*)   (ws + WS_CURS);
    float*  esum   = (float*) (ws + WS_ESUM);
    int*    toke   = (int*)   (ws + WS_TOKE);
    float*  tokw   = (float*) (ws + WS_TOKW);
    int*    btok   = (int*)   (ws + WS_BTOK);
    float*  bw     = (float*) (ws + WS_BW);
    __bf16* xb     = (__bf16*)(ws + WS_XB);
    __bf16* mid    = (__bf16*)(ws + WS_MID);
    __bf16* w1t    = (__bf16*)(ws + WS_W1T);
    __bf16* w2t    = (__bf16*)(ws + WS_W2T);

    (void)hipMemsetAsync(ws, 0, 256, stream);
    (void)hipMemsetAsync(d_out, 0, (size_t)NTOK * ND * sizeof(float), stream);

    gate_kernel<<<NTOK * 32 / 256, 256, 0, stream>>>(x, Wg, toke, tokw, counts);
    prefix_kernel<<<1, 1, 0, stream>>>(counts, offs, curs);
    scatter_kernel<<<NTOK / 256, 256, 0, stream>>>(toke, tokw, curs, btok, bw);
    castx_kernel<<<(NTOK * ND / 4) / 256, 256, 0, stream>>>(x, xb);
    // W1 [E][D][F] -> W1t [E][F][D]; W2 [E][F][D] -> W2t [E][D][F]
    wtrans_kernel<<<dim3(NF / 32, ND / 32, NE), 256, 0, stream>>>(W1, w1t, ND, NF);
    wtrans_kernel<<<dim3(ND / 32, NF / 32, NE), 256, 0, stream>>>(W2, w2t, NF, ND);

    up_kernel<<<dim3(NF / 128, MAXT), 256, 0, stream>>>(xb, w1t, counts, offs, btok, mid);
    down_kernel<<<dim3(ND / 128, MAXT), 256, 0, stream>>>(mid, w2t, counts, offs, btok, bw, y);

    esum_kernel<<<NE, 256, 0, stream>>>(toke, tokw, esum);
    aux_kernel<<<1, 1, 0, stream>>>(esum, y + (size_t)NTOK * ND);
}
